// ActQuantWrapper_88184268522246
// MI455X (gfx1250) — matrix operand order repaired and checked by reading, not compiled, not run on hardware
//
#include <hip/hip_runtime.h>

typedef __attribute__((ext_vector_type(16))) _Float16 v16h;
typedef __attribute__((ext_vector_type(8)))  _Float16 v8h;
typedef __attribute__((ext_vector_type(4)))  _Float16 v4h;
typedef __attribute__((ext_vector_type(8)))  float    v8f;
typedef __attribute__((ext_vector_type(4)))  float    v4f;

#define S_TOK 8192
#define DIM   4096
#define DOUT  4096
#define QPT   15       // (DIM-256)/256 q-features per thread
#define MAXQF 15.0f

#define BM 256
#define BN 128
#define BK 32
#define LDT 40         // padded LDS row stride in halves (80B, 16B aligned)

// ---------- kernel 0: weight f32 -> f16 ----------
__global__ __launch_bounds__(256) void w_to_h(const float* __restrict__ w,
                                              _Float16* __restrict__ wh, int n4) {
  int i = blockIdx.x * blockDim.x + threadIdx.x;
  if (i < n4) {
    v4f f = ((const v4f*)w)[i];
    v4h h;
    h.x = (_Float16)f.x; h.y = (_Float16)f.y;
    h.z = (_Float16)f.z; h.w = (_Float16)f.w;
    ((v4h*)wh)[i] = h;
  }
}

// ---------- kernel 1: per-token 4-bit fake-quant + mix, emit f16 ----------
__global__ __launch_bounds__(256) void quant_x(const float* __restrict__ x,
                                               const int* __restrict__ qidx,
                                               const int* __restrict__ fpidx,
                                               _Float16* __restrict__ xh) {
  __shared__ float smn[256];
  __shared__ float smx[256];
  const int s = blockIdx.x;
  const int t = threadIdx.x;
  const float* xs = x + (size_t)s * DIM;
  _Float16*   xo = xh + (size_t)s * DIM;

  float v[QPT];
  int   id[QPT];
  float mn = 0.0f, mx = 0.0f;   // reference clamps min<=0, max>=0
#pragma unroll
  for (int j = 0; j < QPT; ++j) {
    int k = j * 256 + t;        // coalesced index loads
    int qi = qidx[k];
    id[j] = qi;
    float val = xs[qi];
    v[j] = val;
    mn = fminf(mn, val);
    mx = fmaxf(mx, val);
  }
  smn[t] = mn; smx[t] = mx;
  __syncthreads();
#pragma unroll
  for (int off = 128; off > 0; off >>= 1) {
    if (t < off) {
      smn[t] = fminf(smn[t], smn[t + off]);
      smx[t] = fmaxf(smx[t], smx[t + off]);
    }
    __syncthreads();
  }
  float xmin = smn[0];
  float xmax = smx[0];
  if ((xmin == 0.0f) && (xmax == 0.0f)) { xmin = -1.0f; xmax = 1.0f; }
  float scale = (xmax - xmin) / MAXQF;
  float inv_s = 1.0f / scale;
  float zero  = rintf(-xmin * inv_s);   // rintf: round-half-even == jnp.round

#pragma unroll
  for (int j = 0; j < QPT; ++j) {
    float q  = fminf(fmaxf(rintf(v[j] * inv_s) + zero, 0.0f), MAXQF);
    xo[id[j]] = (_Float16)(scale * (q - zero));
  }
  int fi = fpidx[t];
  xo[fi] = (_Float16)xs[fi];
}

// ---------- kernel 2: f16 WMMA GEMM  out = Xh * Wh^T + bias ----------
// Block tile 256x128, 8 waves in 4(M) x 2(N), each wave owns 64x64 = 4x4 WMMA tiles.
// Double-buffered LDS; next slab prefetched with GLOBAL_LOAD_ASYNC_TO_LDS_B128
// (ASYNCcnt) so no VGPR staging and no loadcnt round-trips in the hot loop.
__global__ __launch_bounds__(256, 1) void gemm_f16(const _Float16* __restrict__ Ah,
                                                   const _Float16* __restrict__ Bh,
                                                   const float* __restrict__ bias,
                                                   float* __restrict__ out) {
  __shared__ __align__(16) _Float16 As[2][BM * LDT];   // 40960 B
  __shared__ __align__(16) _Float16 Bs[2][BN * LDT];   // 20480 B

  const int tid   = threadIdx.x;
  const int lane  = tid & 31;
  const int wave  = tid >> 5;     // 0..7
  const int waveM = wave & 3;     // 4 waves tile M (4*64 = 256)
  const int waveN = wave >> 2;    // 2 waves tile N (2*64 = 128)
  const int mBase = blockIdx.y * BM;
  const int nBase = blockIdx.x * BN;

  v8f acc[4][4];
#pragma unroll
  for (int mt = 0; mt < 4; ++mt)
#pragma unroll
    for (int nt = 0; nt < 4; ++nt)
      acc[mt][nt] = (v8f){0.f, 0.f, 0.f, 0.f, 0.f, 0.f, 0.f, 0.f};

  union U16 { v16h v; v8h h[2]; };

  const int mrow = waveM * 64 + (lane & 15);
  const int nrow = waveN * 64 + (lane & 15);
  const int aK   = (lane < 16) ? 0 : 8;    // ISA 16-bit A layout
  const int bK   = (lane < 16) ? 0 : 8;    // ISA 16-bit B layout

  // Async global->LDS staging of one K-slab (A: 1024 16B-chunks, B: 512).
  // GVS form: 64-bit SGPR base + 32-bit per-lane byte offset; VDST = LDS byte addr.
  auto stage = [&](int buf, int k0) {
#pragma unroll
    for (int r = 0; r < 4; ++r) {
      unsigned cid = tid + r * 256u;   // 0..1023
      unsigned row = cid >> 2;         // 0..255
      unsigned c   = cid & 3u;
      unsigned ldsoff = (unsigned)(uintptr_t)(&As[buf][row * LDT + c * 8]);
      unsigned goff   = (unsigned)(((mBase + row) * DIM + k0 + c * 8) * sizeof(_Float16));
      asm volatile("global_load_async_to_lds_b128 %0, %1, %2"
                   :: "v"(ldsoff), "v"(goff), "s"(Ah) : "memory");
    }
#pragma unroll
    for (int r = 0; r < 2; ++r) {
      unsigned cid = tid + r * 256u;   // 0..511
      unsigned row = cid >> 2;         // 0..127
      unsigned c   = cid & 3u;
      unsigned ldsoff = (unsigned)(uintptr_t)(&Bs[buf][row * LDT + c * 8]);
      unsigned goff   = (unsigned)(((nBase + row) * DIM + k0 + c * 8) * sizeof(_Float16));
      asm volatile("global_load_async_to_lds_b128 %0, %1, %2"
                   :: "v"(ldsoff), "v"(goff), "s"(Bh) : "memory");
    }
  };

  stage(0, 0);
  int buf = 0;
  for (int k0 = 0; k0 < DIM; k0 += BK) {
    asm volatile("s_wait_asynccnt 0x0" ::: "memory");  // wave's async fills of `buf` done
    __syncthreads();                                   // all waves: slab visible, reads of buf^1 drained
    if (k0 + BK < DIM) stage(buf ^ 1, k0 + BK);        // async prefetch next slab (overlaps math)

    const _Float16* Ab = As[buf];
    const _Float16* Bb = Bs[buf];

    U16 aF[4];
#pragma unroll
    for (int mt = 0; mt < 4; ++mt) {
      const _Float16* p = Ab + (mrow + mt * 16) * LDT;
      aF[mt].h[0] = *(const v8h*)(p + aK);        // K {0..7}  or {8..15}
      aF[mt].h[1] = *(const v8h*)(p + aK + 16);   // K {16..23} or {24..31}
    }
    U16 bF[4];
#pragma unroll
    for (int nt = 0; nt < 4; ++nt) {
      const _Float16* p = Bb + (nrow + nt * 16) * LDT + bK;
      bF[nt].h[0] = *(const v8h*)(p);             // K {0..7}  or {8..15}
      bF[nt].h[1] = *(const v8h*)(p + 16);        // K {16..23} or {24..31}
    }

#pragma unroll
    for (int nt = 0; nt < 4; ++nt)
#pragma unroll
      for (int mt = 0; mt < 4; ++mt)
        acc[mt][nt] = __builtin_amdgcn_wmma_f32_16x16x32_f16(
            /*neg_a=*/false, aF[mt].v, /*neg_b=*/false, bF[nt].v,
            /*c_mod=*/(short)0, acc[mt][nt],
            /*reuse_a=*/false, /*reuse_b=*/false);
    buf ^= 1;
  }

  // Epilogue: D layout — VGPR r: M = r (lanes 0-15) / 8+r (lanes 16-31); N = lane%16
  const int mOff  = (lane < 16) ? 0 : 8;
  const int nCol0 = nBase + waveN * 64 + (lane & 15);
#pragma unroll
  for (int nt = 0; nt < 4; ++nt) {
    int col = nCol0 + nt * 16;
    float b = bias[col];
#pragma unroll
    for (int mt = 0; mt < 4; ++mt) {
      int mrow0 = mBase + waveM * 64 + mt * 16 + mOff;
#pragma unroll
      for (int r = 0; r < 8; ++r)
        out[(size_t)(mrow0 + r) * DOUT + col] = acc[mt][nt][r] + b;
    }
  }
}

extern "C" void kernel_launch(void* const* d_in, const int* in_sizes, int n_in,
                              void* d_out, int out_size, void* d_ws, size_t ws_size,
                              hipStream_t stream) {
  const float* x     = (const float*)d_in[0];
  const float* w     = (const float*)d_in[1];
  const float* bias  = (const float*)d_in[2];
  const int*   qidx  = (const int*)d_in[3];
  const int*   fpidx = (const int*)d_in[4];
  float* out = (float*)d_out;

  _Float16* xh = (_Float16*)d_ws;                 // 8192*4096 f16 = 64 MB
  _Float16* wh = xh + (size_t)S_TOK * DIM;        // 4096*4096 f16 = 32 MB

  int n4 = (DOUT * DIM) / 4;
  w_to_h<<<(n4 + 255) / 256, 256, 0, stream>>>(w, wh, n4);
  quant_x<<<S_TOK, 256, 0, stream>>>(x, qidx, fpidx, xh);

  dim3 grid(DOUT / BN, S_TOK / BM);
  gemm_f16<<<grid, 256, 0, stream>>>(xh, wh, bias, out);
}
